// BraidCausalSelfAttention_24292335026389
// MI455X (gfx1250) — matrix operand order repaired and checked by reading, not compiled, not run on hardware
//
#include <hip/hip_runtime.h>

// ---------------- types ----------------
typedef __attribute__((ext_vector_type(16))) __bf16 v16bf;
typedef __attribute__((ext_vector_type(8)))  float  v8f;
typedef __attribute__((ext_vector_type(8)))  __bf16 bf16x8;
typedef __attribute__((ext_vector_type(4)))  __bf16 bf16x4;
typedef __attribute__((ext_vector_type(4)))  unsigned int u32x4;
typedef __attribute__((ext_vector_type(8)))  int i32x8;
typedef __attribute__((ext_vector_type(4)))  int i32x4;

#define TSEQ 2048
#define CDIM 1024
#define NB   2

#if defined(__has_builtin)
#if __has_builtin(__builtin_amdgcn_tensor_load_to_lds) && \
    __has_builtin(__builtin_amdgcn_s_wait_tensorcnt)
#define USE_TDM 1
#endif
#endif
#ifndef USE_TDM
#define USE_TDM 0
#endif

// Padded x-tile LDS layout: pad 16B after every 1024B of stream
// row stride = 2048 + 2*16 = 2080 bytes  (520 DW -> +8 banks per row)
#define XROW_BYTES 2080
__device__ __forceinline__ unsigned lds_x_off(int m, int k) {
  return (unsigned)(m * XROW_BYTES + 2 * k + ((k >> 9) << 4));
}

#if USE_TDM
// 2-D TDM global->LDS copy of a bf16 tile (data_size = 2 bytes).
// tile_w elems per row, tile_h rows; row stride in elems; tensor dims for OOB.
__device__ __forceinline__ void tdm_load_2d_bf16(
    unsigned lds_byte_addr, const void* gaddr,
    unsigned tile_w, unsigned tile_h,
    unsigned row_stride_elems, unsigned tensor_w, unsigned tensor_h,
    bool pad) {
  unsigned long long ga = (unsigned long long)gaddr;
  u32x4 g0;
  g0[0] = 1u;                                            // count=1, user desc
  g0[1] = lds_byte_addr;                                 // lds_addr
  g0[2] = (unsigned)(ga & 0xFFFFFFFFu);                  // global_addr[31:0]
  g0[3] = (unsigned)((ga >> 32) & 0x01FFFFFFu) | (2u << 30);  // [56:32], type=2
  i32x8 g1;
  unsigned d0 = (1u << 16);                              // data_size=1 (2B)
  if (pad) d0 |= (1u << 20) | (7u << 22) | (3u << 25);   // pad: 256DW int, 4DW amt
  g1[0] = (int)d0;
  g1[1] = (int)((tensor_w & 0xFFFFu) << 16);             // tensor_dim0[15:0]
  g1[2] = (int)((tensor_w >> 16) | ((tensor_h & 0xFFFFu) << 16));
  g1[3] = (int)((tensor_h >> 16) | ((tile_w & 0xFFFFu) << 16));  // tile_dim0
  g1[4] = (int)(tile_h & 0xFFFFu);                       // tile_dim1 (tile_dim2=0)
  g1[5] = (int)row_stride_elems;                         // dim0_stride[31:0]
  g1[6] = 0;                                             // dim0_stride[47:32], dim1_stride lo
  g1[7] = 0;
  i32x4 z4 = {0, 0, 0, 0};
  i32x8 z8 = {0, 0, 0, 0, 0, 0, 0, 0};
  __builtin_amdgcn_tensor_load_to_lds(g0, g1, z4, z4, z8, 0);
}
#endif

// ---------------- f32 -> bf16 convert ----------------
__global__ void cvt_f32_to_bf16(const float* __restrict__ src,
                                __bf16* __restrict__ dst, int n) {
  for (int i = blockIdx.x * blockDim.x + threadIdx.x; i < n;
       i += gridDim.x * blockDim.x) {
    dst[i] = (__bf16)src[i];
  }
}

// ---------------- fused QKV GEMM + rope + rmsnorm + braid-dot ----------------
// grid: (256, 32)  block: 128 (4 waves)
__global__ __launch_bounds__(128)
void qkv_rope_norm_braid(const __bf16* __restrict__ xb,
                         const __bf16* __restrict__ wqkv,
                         const float* __restrict__ cosp,
                         const float* __restrict__ sinp,
                         const float* __restrict__ wb,
                         float* __restrict__ sq,
                         float* __restrict__ sk,
                         __bf16* __restrict__ vb) {
  __shared__ __align__(16) unsigned char lxraw[16 * XROW_BYTES];  // padded 16x1024 bf16
  __shared__ float lo[16][68];

  const int tid  = threadIdx.x;
  const int lane = tid & 31;
  const int wave = tid >> 5;
  const int m16  = lane & 15;
  const int hi   = lane >> 4;
  const int slot = blockIdx.y;
  const int row0 = blockIdx.x * 16;

  const int ncol = slot * 64 + wave * 16 + m16;
  const __bf16* wrow = wqkv + (size_t)ncol * CDIM;
  const __bf16* xtile = xb + (size_t)row0 * CDIM;

#if USE_TDM
  if (wave == 0) {
    tdm_load_2d_bf16((unsigned)(unsigned long long)(uintptr_t)&lxraw[0], xtile,
                     CDIM, 16, CDIM, CDIM, 16, /*pad=*/true);
    __builtin_amdgcn_s_wait_tensorcnt(0);
  }
  __syncthreads();
#else
  for (int e = tid * 8; e < 16 * CDIM; e += 128 * 8) {
    int mr = e >> 10, cc = e & 1023;
    bf16x8 t = *(const bf16x8*)&xtile[(size_t)mr * CDIM + cc];
#pragma unroll
    for (int j = 0; j < 8; ++j)
      *(__bf16*)&lxraw[lds_x_off(mr, cc + j)] = t[j];
  }
  __syncthreads();
#endif

  v8f acc = {0.f, 0.f, 0.f, 0.f, 0.f, 0.f, 0.f, 0.f};

  for (int kk = 0; kk < CDIM; kk += 32) {
    // A fragment from padded LDS tile
    v16bf a;
#pragma unroll
    for (int i = 0; i < 8; ++i) {
      int k0 = kk + ((i < 4) ? (2 * i + 8 * hi) : (16 + 2 * (i - 4) + 8 * hi));
      a[2 * i]     = *(const __bf16*)&lxraw[lds_x_off(m16, k0)];
      a[2 * i + 1] = *(const __bf16*)&lxraw[lds_x_off(m16, k0 + 1)];
    }
    // B fragment: K = 8*hi..+7 and 16+8*hi..+7 of weight row ncol
    bf16x8 b0 = *(const bf16x8*)&wrow[kk + 8 * hi];
    bf16x8 b1 = *(const bf16x8*)&wrow[kk + 16 + 8 * hi];
    v16bf b;
#pragma unroll
    for (int i = 0; i < 8; ++i) { b[i] = b0[i]; b[i + 8] = b1[i]; }

    __builtin_prefetch(&wrow[kk + 64], 0, 1);

    acc = __builtin_amdgcn_wmma_f32_16x16x32_bf16(
        false, a, false, b, (short)0, acc, false, false);
  }

#pragma unroll
  for (int j = 0; j < 8; ++j) lo[j + 8 * hi][wave * 16 + m16] = acc[j];
  __syncthreads();

  if (slot < 24) {
    if (tid < 16) {
      const int m = tid;
      const int r = row0 + m;
      const int bidx = r >> 11;
      const int t    = r & (TSEQ - 1);
      float sumsq = 0.f, dot = 0.f;
#pragma unroll 4
      for (int d = 0; d < 32; ++d) {
        float x1 = lo[m][d], x2 = lo[m][d + 32];
        float co = cosp[t * 32 + d], si = sinp[t * 32 + d];
        float r1 = x1 * co + x2 * si;
        float r2 = x2 * co - x1 * si;
        sumsq += r1 * r1 + r2 * r2;
        dot   += r1 * wb[d] + r2 * wb[d + 32];
      }
      float s = dot * rsqrtf(sumsq * (1.f / 64.f) + 1e-6f);
      if (slot < 16) sq[((size_t)bidx * 16 + slot) * TSEQ + t] = s;
      else           sk[((size_t)bidx * 8 + (slot - 16)) * TSEQ + t] = s;
    }
  } else {
    const int vh = slot - 24;
#pragma unroll
    for (int j = 0; j < 8; ++j) {
      int e = tid * 8 + j;
      int m = e >> 6, n = e & 63;
      int r = row0 + m;
      int bidx = r >> 11, t = r & (TSEQ - 1);
      vb[(((size_t)bidx * 8 + vh) * TSEQ + t) * 64 + n] = (__bf16)lo[m][n];
    }
  }
}

// ---------------- braid attention: y = sigmoid(s_q+s_k) @ v ----------------
// grid: (T/16, 16 heads, B)  block: 128; TDM double-buffered v chunks
__global__ __launch_bounds__(128)
void braid_attn(const float* __restrict__ sq, const float* __restrict__ sk,
                const __bf16* __restrict__ vb, __bf16* __restrict__ yb) {
  __shared__ __align__(16) __bf16 lv[2 * 32 * 64];

  const int tid  = threadIdx.x;
  const int lane = tid & 31;
  const int wave = tid >> 5;
  const int m16  = lane & 15;
  const int hi   = lane >> 4;
  const int t0   = blockIdx.x * 16;
  const int h    = blockIdx.y;
  const int b    = blockIdx.z;
  const int kvh  = h >> 1;

  const float*  sqp = sq + ((size_t)b * 16 + h) * TSEQ;
  const float*  skp = sk + ((size_t)b * 8 + kvh) * TSEQ;
  const __bf16* vbp = vb + ((size_t)b * 8 + kvh) * TSEQ * 64;

  const int   trow = t0 + m16;
  const float sqv  = sqp[trow];

  v8f acc = {0.f, 0.f, 0.f, 0.f, 0.f, 0.f, 0.f, 0.f};
  const int smax = t0 + 16;

#if USE_TDM
  if (wave == 0) {
    tdm_load_2d_bf16((unsigned)(unsigned long long)(uintptr_t)&lv[0],
                     vbp, 64, 32, 64, 64, (unsigned)TSEQ, /*pad=*/false);
  }
#endif

  int buf = 0;
  for (int s0 = 0; s0 < smax; s0 += 32) {
    const int has_next = (s0 + 32 < smax);
#if USE_TDM
    if (wave == 0) {
      if (has_next) {
        tdm_load_2d_bf16(
            (unsigned)(unsigned long long)(uintptr_t)&lv[(buf ^ 1) * 2048],
            vbp + (size_t)(s0 + 32) * 64, 64, 32, 64, 64,
            (unsigned)(TSEQ - (s0 + 32)), /*pad=*/false);
        __builtin_amdgcn_s_wait_tensorcnt(1);  // current chunk resident
      } else {
        __builtin_amdgcn_s_wait_tensorcnt(0);
      }
    }
    __syncthreads();
#else
    __syncthreads();
    {
      int e  = tid * 16;
      int sr = e >> 6, sc = e & 63;
      int srow = s0 + sr; if (srow > TSEQ - 1) srow = TSEQ - 1;
      *(bf16x8*)&lv[buf * 2048 + sr * 64 + sc] =
          *(const bf16x8*)&vbp[(size_t)srow * 64 + sc];
      *(bf16x8*)&lv[buf * 2048 + sr * 64 + sc + 8] =
          *(const bf16x8*)&vbp[(size_t)srow * 64 + sc + 8];
    }
    __syncthreads();
#endif
    const __bf16* lvb = &lv[buf * 2048];

    // A fragment = sigmoid scores (exact 0.0 where masked)
    v16bf a;
#pragma unroll
    for (int i = 0; i < 8; ++i) {
      int k0 = (i < 4) ? (2 * i + 8 * hi) : (16 + 2 * (i - 4) + 8 * hi);
      int s_a = s0 + k0;
      int s_b = s0 + k0 + 1;
      int c_a = (s_a > TSEQ - 1) ? TSEQ - 1 : s_a;
      int c_b = (s_b > TSEQ - 1) ? TSEQ - 1 : s_b;
      float ska = skp[c_a];
      float skb = skp[c_b];
      float va  = (s_a <= trow) ? 1.f / (1.f + __expf(-(sqv + ska))) : 0.f;
      float vb2 = (s_b <= trow) ? 1.f / (1.f + __expf(-(sqv + skb))) : 0.f;
      a[2 * i]     = (__bf16)va;
      a[2 * i + 1] = (__bf16)vb2;
    }
    // B fragment from LDS
    v16bf bfr;
    const int ncolv = wave * 16 + m16;
#pragma unroll
    for (int i = 0; i < 16; ++i) bfr[i] = lvb[((i < 8) ? (8 * hi + i) : (16 + 8 * hi + i - 8)) * 64 + ncolv];

    acc = __builtin_amdgcn_wmma_f32_16x16x32_bf16(
        false, a, false, bfr, (short)0, acc, false, false);

    __syncthreads();   // all waves done with lv[buf] before it is overwritten
    buf ^= 1;
  }

  const float scale = 1.f / (45.254833995939045f + 1e-6f);
#pragma unroll
  for (int j = 0; j < 8; ++j) {
    int m = j + 8 * hi;
    int t = t0 + m;
    yb[((size_t)b * TSEQ + t) * CDIM + h * 64 + wave * 16 + m16] =
        (__bf16)(acc[j] * scale);
  }
}

// ---------------- output projection: out = y @ Wproj^T (f32 out) ----------------
// grid: (256, 16)  block: 128
__global__ __launch_bounds__(128)
void proj_gemm(const __bf16* __restrict__ yb, const __bf16* __restrict__ wpb,
               float* __restrict__ out) {
  __shared__ __align__(16) unsigned char lxraw[16 * XROW_BYTES];

  const int tid  = threadIdx.x;
  const int lane = tid & 31;
  const int wave = tid >> 5;
  const int m16  = lane & 15;
  const int hi   = lane >> 4;
  const int row0 = blockIdx.x * 16;
  const int ncol = blockIdx.y * 64 + wave * 16 + m16;
  const __bf16* wrow = wpb + (size_t)ncol * CDIM;
  const __bf16* ytile = yb + (size_t)row0 * CDIM;

#if USE_TDM
  if (wave == 0) {
    tdm_load_2d_bf16((unsigned)(unsigned long long)(uintptr_t)&lxraw[0], ytile,
                     CDIM, 16, CDIM, CDIM, 16, /*pad=*/true);
    __builtin_amdgcn_s_wait_tensorcnt(0);
  }
  __syncthreads();
#else
  for (int e = tid * 8; e < 16 * CDIM; e += 128 * 8) {
    int mr = e >> 10, cc = e & 1023;
    bf16x8 t = *(const bf16x8*)&ytile[(size_t)mr * CDIM + cc];
#pragma unroll
    for (int j = 0; j < 8; ++j)
      *(__bf16*)&lxraw[lds_x_off(mr, cc + j)] = t[j];
  }
  __syncthreads();
#endif

  v8f acc = {0.f, 0.f, 0.f, 0.f, 0.f, 0.f, 0.f, 0.f};

  for (int kk = 0; kk < CDIM; kk += 32) {
    v16bf a;
#pragma unroll
    for (int i = 0; i < 8; ++i) {
      int k0 = kk + ((i < 4) ? (2 * i + 8 * hi) : (16 + 2 * (i - 4) + 8 * hi));
      a[2 * i]     = *(const __bf16*)&lxraw[lds_x_off(m16, k0)];
      a[2 * i + 1] = *(const __bf16*)&lxraw[lds_x_off(m16, k0 + 1)];
    }
    bf16x8 b0 = *(const bf16x8*)&wrow[kk + 8 * hi];
    bf16x8 b1 = *(const bf16x8*)&wrow[kk + 16 + 8 * hi];
    v16bf b;
#pragma unroll
    for (int i = 0; i < 8; ++i) { b[i] = b0[i]; b[i + 8] = b1[i]; }

    __builtin_prefetch(&wrow[kk + 64], 0, 1);

    acc = __builtin_amdgcn_wmma_f32_16x16x32_bf16(
        false, a, false, b, (short)0, acc, false, false);
  }

#pragma unroll
  for (int j = 0; j < 8; ++j) {
    int m = j + 8 * hi;
    out[(size_t)(row0 + m) * CDIM + ncol] = acc[j];
  }
}

// ---------------- host launcher ----------------
extern "C" void kernel_launch(void* const* d_in, const int* in_sizes, int n_in,
                              void* d_out, int out_size, void* d_ws,
                              size_t ws_size, hipStream_t stream) {
  (void)in_sizes; (void)n_in; (void)out_size; (void)ws_size;

  const float* x    = (const float*)d_in[0];
  const float* cosp = (const float*)d_in[1];
  const float* sinp = (const float*)d_in[2];
  const float* Wq   = (const float*)d_in[3];
  const float* Wk   = (const float*)d_in[4];
  const float* Wv   = (const float*)d_in[5];
  const float* Wp   = (const float*)d_in[6];
  const float* wb   = (const float*)d_in[7];

  constexpr size_t N_X    = (size_t)NB * TSEQ * CDIM;
  constexpr size_t N_WQKV = 2048u * CDIM;
  constexpr size_t N_WP   = 1024u * CDIM;
  constexpr size_t N_SQ   = (size_t)NB * 16 * TSEQ;
  constexpr size_t N_SK   = (size_t)NB * 8 * TSEQ;
  constexpr size_t N_V    = (size_t)NB * 8 * TSEQ * 64;

  char* ws = (char*)d_ws;
  size_t off = 0;
  __bf16* xb   = (__bf16*)(ws + off); off += N_X * 2;
  __bf16* wqkv = (__bf16*)(ws + off); off += N_WQKV * 2;
  __bf16* wpb  = (__bf16*)(ws + off); off += N_WP * 2;
  float*  sq   = (float*)(ws + off);  off += N_SQ * 4;
  float*  sk   = (float*)(ws + off);  off += N_SK * 4;
  __bf16* vbuf = (__bf16*)(ws + off); off += N_V * 2;
  __bf16* ybuf = (__bf16*)(ws + off); off += N_X * 2;

  cvt_f32_to_bf16<<<dim3(1024), dim3(256), 0, stream>>>(x, xb, (int)N_X);
  cvt_f32_to_bf16<<<dim3(512),  dim3(256), 0, stream>>>(Wq, wqkv, 1048576);
  cvt_f32_to_bf16<<<dim3(256),  dim3(256), 0, stream>>>(Wk, wqkv + 1048576, 524288);
  cvt_f32_to_bf16<<<dim3(256),  dim3(256), 0, stream>>>(Wv, wqkv + 1572864, 524288);
  cvt_f32_to_bf16<<<dim3(512),  dim3(256), 0, stream>>>(Wp, wpb, (int)N_WP);

  qkv_rope_norm_braid<<<dim3((NB * TSEQ) / 16, 32), dim3(128), 0, stream>>>(
      xb, wqkv, cosp, sinp, wb, sq, sk, vbuf);

  braid_attn<<<dim3(TSEQ / 16, 16, NB), dim3(128), 0, stream>>>(
      sq, sk, vbuf, ybuf);

  proj_gemm<<<dim3((NB * TSEQ) / 16, 16), dim3(128), 0, stream>>>(
      ybuf, wpb, (float*)d_out);
}
